// MultiHeadAttention_3891240370897
// MI455X (gfx1250) — matrix operand order repaired and checked by reading, not compiled, not run on hardware
//
#include <hip/hip_runtime.h>

// ---------------------------------------------------------------------------
// Types for CDNA5 WMMA (wave32, V_WMMA_F32_16X16X32_BF16)
// ---------------------------------------------------------------------------
typedef __bf16 bf16;
typedef __attribute__((ext_vector_type(16))) __bf16 v16bf;
typedef __attribute__((ext_vector_type(8)))  __bf16 bf16x8;
typedef __attribute__((ext_vector_type(8)))  float  v8f;
typedef __attribute__((ext_vector_type(4)))  int    v4i;

#define WMMA_BF16(a, b, c) \
  __builtin_amdgcn_wmma_f32_16x16x32_bf16(false, (a), false, (b), (short)0, (c), false, false)

// --- async global->LDS path (GLOBAL_LOAD_ASYNC_TO_LDS_B128, ASYNCcnt) ------
#if defined(__has_builtin)
#if __has_builtin(__builtin_amdgcn_global_load_async_to_lds_b128) && \
    __has_builtin(__builtin_amdgcn_s_wait_asynccnt)
#define HAVE_ASYNC_LDS 1
#pragma message("CDNA5: using GLOBAL_LOAD_ASYNC_TO_LDS_B128 path")
#endif
#endif
#ifndef HAVE_ASYNC_LDS
#define HAVE_ASYNC_LDS 0
#pragma message("CDNA5: async-to-LDS builtins NOT available, sync fallback")
#endif

__device__ __forceinline__ void copy16_to_lds(bf16* lds, const bf16* g) {
#if HAVE_ASYNC_LDS
  __builtin_amdgcn_global_load_async_to_lds_b128(
      (__attribute__((address_space(1))) v4i*)g,
      (__attribute__((address_space(3))) v4i*)lds, 0, 0);
#else
  *(bf16x8*)lds = *(const bf16x8*)g;
#endif
}
__device__ __forceinline__ void async_wait0() {
#if HAVE_ASYNC_LDS
  __builtin_amdgcn_s_wait_asynccnt(0);
#endif
}

__device__ __forceinline__ bf16 f2bf(float f) {
  unsigned u = __builtin_bit_cast(unsigned, f);
  u += 0x7FFFu + ((u >> 16) & 1u);           // round-to-nearest-even
  unsigned short s = (unsigned short)(u >> 16);
  return __builtin_bit_cast(bf16, s);
}

__device__ __forceinline__ v16bf mk16(bf16x8 lo, bf16x8 hi) {
  return __builtin_shufflevector(lo, hi, 0,1,2,3,4,5,6,7,8,9,10,11,12,13,14,15);
}

// Problem constants (fixed by the reference)
constexpr int DM   = 1024;     // d_model
constexpr int NH   = 16;       // heads
constexpr int DK   = 64;       // head dim
constexpr int BS   = 4;        // batch
constexpr int SL   = 2048;     // seq len
constexpr int MT   = BS * SL;  // 8192 rows

// ---------------------------------------------------------------------------
// fp32 -> bf16 cast
// ---------------------------------------------------------------------------
__global__ void cast_f32_bf16(const float* __restrict__ s, bf16* __restrict__ d, int n) {
  int i = blockIdx.x * 256 + threadIdx.x;
  if (i < n) d[i] = f2bf(s[i]);
}

// ---------------------------------------------------------------------------
// GEMM: C[M=8192, N=1024] = A[M,1024] * W[N,1024]^T + bias
// A row-major [m][k] bf16, W row-major [n][k] bf16 (torch Linear layout).
// Block tile 128x128, K-step 32, 8 waves (4x2), each wave 32x64 = 2x4 C tiles.
// Double-buffered LDS tiles filled by async global->LDS DMA.
// OUT_F32==0: store bf16 (QKV path). OUT_F32==1: store fp32 (final output).
// ---------------------------------------------------------------------------
template <int OUT_F32>
__global__ void __launch_bounds__(256)
gemm_bf16(const bf16* __restrict__ A, const bf16* __restrict__ W,
          const float* __restrict__ bias, void* __restrict__ Cout) {
  constexpr int LDT = 40;                       // padded row stride (halfs), 80B
  __shared__ __align__(16) bf16 As[2][128 * LDT];
  __shared__ __align__(16) bf16 Bs[2][128 * LDT];

  const int tid  = threadIdx.x;
  const int wave = tid >> 5;
  const int lane = tid & 31;
  const int l15  = lane & 15;
  const int hi   = lane >> 4;
  const int wm   = wave >> 1;                   // 0..3
  const int wn   = wave & 1;                    // 0..1
  const int m0   = blockIdx.y * 128;
  const int n0   = blockIdx.x * 128;

  // cooperative tile fill: 512 chunks of 8 halfs each, 2 per thread
  const int r_a = tid >> 2;                     // 0..63 (+64 for p=1)
  const int cc  = (tid & 3) * 8;                // 0..24

  v8f c[2][4];
#pragma unroll
  for (int mt = 0; mt < 2; ++mt)
#pragma unroll
    for (int nt = 0; nt < 4; ++nt)
      c[mt][nt] = (v8f){0.f,0.f,0.f,0.f,0.f,0.f,0.f,0.f};

  // prologue: fill buffer 0
#pragma unroll
  for (int p = 0; p < 2; ++p) {
    int r = r_a + p * 64;
    copy16_to_lds(&As[0][r * LDT + cc], &A[(size_t)(m0 + r) * DM + cc]);
    copy16_to_lds(&Bs[0][r * LDT + cc], &W[(size_t)(n0 + r) * DM + cc]);
  }

  for (int it = 0; it < DM / 32; ++it) {
    async_wait0();
    __syncthreads();
    // prefetch next tile into the other buffer while we compute
    if (it + 1 < DM / 32) {
      int k0n = (it + 1) * 32;
      int nb  = (it + 1) & 1;
#pragma unroll
      for (int p = 0; p < 2; ++p) {
        int r = r_a + p * 64;
        copy16_to_lds(&As[nb][r * LDT + cc], &A[(size_t)(m0 + r) * DM + k0n + cc]);
        copy16_to_lds(&Bs[nb][r * LDT + cc], &W[(size_t)(n0 + r) * DM + k0n + cc]);
      }
    }

    const bf16* as = As[it & 1];
    const bf16* bs = Bs[it & 1];
    v16bf a[2], b[4];
#pragma unroll
    for (int mt = 0; mt < 2; ++mt) {
      const bf16* ap = &as[(wm * 32 + mt * 16 + l15) * LDT];
      a[mt] = mk16(*(const bf16x8*)(ap + 8 * hi), *(const bf16x8*)(ap + 16 + 8 * hi));
    }
#pragma unroll
    for (int nt = 0; nt < 4; ++nt) {
      const bf16* bp = &bs[(wn * 64 + nt * 16 + l15) * LDT];
      b[nt] = mk16(*(const bf16x8*)(bp + 8 * hi), *(const bf16x8*)(bp + 16 + 8 * hi));
    }
#pragma unroll
    for (int mt = 0; mt < 2; ++mt)
#pragma unroll
      for (int nt = 0; nt < 4; ++nt)
        c[mt][nt] = WMMA_BF16(a[mt], b[nt], c[mt][nt]);
  }

  // epilogue: C layout -> row = 8*hi + v (VGPR v), col = lane&15 per 16x16 tile
#pragma unroll
  for (int mt = 0; mt < 2; ++mt) {
#pragma unroll
    for (int nt = 0; nt < 4; ++nt) {
      int col  = n0 + wn * 64 + nt * 16 + l15;
      float bv = bias[col];
#pragma unroll
      for (int v = 0; v < 8; ++v) {
        int row   = m0 + wm * 32 + mt * 16 + 8 * hi + v;
        float val = c[mt][nt][v] + bv;
        if (OUT_F32) ((float*)Cout)[(size_t)row * DM + col] = val;
        else         ((bf16*) Cout)[(size_t)row * DM + col] = f2bf(val);
      }
    }
  }
}

// ---------------------------------------------------------------------------
// Flash attention: block = 128 query rows of one (b,h); 8 waves x 16 rows.
// Q/K/V stored as [b*2048 + l][h*64 + d] bf16. Key-padding mask per (b, j).
// Online softmax, K-tile of 64 keys per iteration. K tile via async DMA.
// ---------------------------------------------------------------------------
__global__ void __launch_bounds__(256)
attn_kernel(const bf16* __restrict__ Q, const bf16* __restrict__ K,
            const bf16* __restrict__ V, const int* __restrict__ mask,
            bf16* __restrict__ O) {
  constexpr int LDK = 72;                       // padded row stride (halfs), 144B
  __shared__ __align__(16) bf16 Kt[64 * LDK];   // [j][d]
  __shared__ __align__(16) bf16 Vt[64 * LDK];   // transposed: [d][j]
  __shared__ __align__(16) bf16 Ps[8 * 16 * LDK];  // per-wave P tile [16][64]

  const int tid  = threadIdx.x;
  const int wave = tid >> 5;
  const int lane = tid & 31;
  const int l15  = lane & 15;
  const int hi   = lane >> 4;
  const int b    = blockIdx.y >> 4;
  const int h    = blockIdx.y & 15;
  const int q0   = blockIdx.x * 128 + wave * 16;
  const float scale = 0.125f;                   // 1/sqrt(64)
  const size_t hoff = (size_t)h * DK;

  // Q fragments (A layout): row = lane&15, K-chunks at 8*hi and 16+8*hi
  v16bf qf[2];
#pragma unroll
  for (int kk = 0; kk < 2; ++kk) {
    const bf16* qp = Q + (size_t)(b * SL + q0 + l15) * DM + hoff + kk * 32;
    qf[kk] = mk16(*(const bf16x8*)(qp + 8 * hi), *(const bf16x8*)(qp + 16 + 8 * hi));
  }

  float m_r[8], l_r[8];
  v8f o[4];
#pragma unroll
  for (int v = 0; v < 8; ++v) { m_r[v] = -1e30f; l_r[v] = 0.f; }
#pragma unroll
  for (int nt = 0; nt < 4; ++nt) o[nt] = (v8f){0.f,0.f,0.f,0.f,0.f,0.f,0.f,0.f};

  bf16* Pw = &Ps[wave * 16 * LDK];

  for (int j0 = 0; j0 < SL; j0 += 64) {
    __syncthreads();
    // load K tile (async DMA) and transposed V tile: 512 chunks of 8 halfs
#pragma unroll
    for (int p = 0; p < 2; ++p) {
      int ch = tid + p * 256;
      int r  = ch >> 3;                         // key row 0..63
      int d  = (ch & 7) * 8;                    // 0..56
      size_t g = (size_t)(b * SL + j0 + r) * DM + hoff + d;
      copy16_to_lds(&Kt[r * LDK + d], &K[g]);
      bf16x8 vv = *(const bf16x8*)&V[g];
#pragma unroll
      for (int e = 0; e < 8; ++e) Vt[(d + e) * LDK + r] = vv[e];
    }
    async_wait0();
    __syncthreads();

    // S = Q * K^T   (B-frag: N=j from Kt row, K=d chunks at 8*hi and 16+8*hi)
    v8f s[4];
#pragma unroll
    for (int nt = 0; nt < 4; ++nt) {
      v8f acc = (v8f){0.f,0.f,0.f,0.f,0.f,0.f,0.f,0.f};
#pragma unroll
      for (int kk = 0; kk < 2; ++kk) {
        const bf16* kp = &Kt[(nt * 16 + l15) * LDK + kk * 32];
        v16bf bb = mk16(*(const bf16x8*)(kp + 8 * hi), *(const bf16x8*)(kp + 16 + 8 * hi));
        acc = WMMA_BF16(qf[kk], bb, acc);
      }
      s[nt] = acc;
    }

    // scale + key-padding mask (reference: logits*(1-m) + m*(-1e30))
#pragma unroll
    for (int nt = 0; nt < 4; ++nt) {
      int mk = mask[b * SL + j0 + nt * 16 + l15];
#pragma unroll
      for (int v = 0; v < 8; ++v)
        s[nt][v] = mk ? -1e30f : s[nt][v] * scale;
    }

    // online softmax (row = 8*hi + v lives in 16-lane half-wave)
    float alpha[8];
#pragma unroll
    for (int v = 0; v < 8; ++v) {
      float r = s[0][v];
#pragma unroll
      for (int nt = 1; nt < 4; ++nt) r = fmaxf(r, s[nt][v]);
      for (int sh = 1; sh < 16; sh <<= 1) r = fmaxf(r, __shfl_xor(r, sh, 32));
      float mn = fmaxf(m_r[v], r);
      alpha[v] = __expf(m_r[v] - mn);
      m_r[v]   = mn;
    }
    float rs[8];
#pragma unroll
    for (int v = 0; v < 8; ++v) rs[v] = 0.f;
#pragma unroll
    for (int nt = 0; nt < 4; ++nt)
#pragma unroll
      for (int v = 0; v < 8; ++v) {
        float p = __expf(s[nt][v] - m_r[v]);
        s[nt][v] = p;
        rs[v] += p;
      }
#pragma unroll
    for (int v = 0; v < 8; ++v) {
      for (int sh = 1; sh < 16; sh <<= 1) rs[v] += __shfl_xor(rs[v], sh, 32);
      l_r[v] = l_r[v] * alpha[v] + rs[v];
    }
#pragma unroll
    for (int nt = 0; nt < 4; ++nt)
#pragma unroll
      for (int v = 0; v < 8; ++v) o[nt][v] *= alpha[v];

    // write P (C layout) -> LDS [i][j], then reload as A-fragments
#pragma unroll
    for (int nt = 0; nt < 4; ++nt)
#pragma unroll
      for (int v = 0; v < 8; ++v)
        Pw[(8 * hi + v) * LDK + nt * 16 + l15] = f2bf(s[nt][v]);

    // O += P * V  (A-frag from Pw, B-frag: N=d from Vt row, K=j chunks at 8*hi and 16+8*hi)
#pragma unroll
    for (int kk = 0; kk < 2; ++kk) {
      const bf16* pp = &Pw[l15 * LDK + kk * 32];
      v16bf pa = mk16(*(const bf16x8*)(pp + 8 * hi), *(const bf16x8*)(pp + 16 + 8 * hi));
#pragma unroll
      for (int nt = 0; nt < 4; ++nt) {
        const bf16* vp = &Vt[(nt * 16 + l15) * LDK + kk * 32];
        v16bf vb = mk16(*(const bf16x8*)(vp + 8 * hi), *(const bf16x8*)(vp + 16 + 8 * hi));
        o[nt] = WMMA_BF16(pa, vb, o[nt]);
      }
    }
  }

  // normalize + store (layout matches transpose(0,2,1,3).reshape concat-heads)
#pragma unroll
  for (int v = 0; v < 8; ++v) l_r[v] = 1.0f / l_r[v];
#pragma unroll
  for (int nt = 0; nt < 4; ++nt)
#pragma unroll
    for (int v = 0; v < 8; ++v) {
      int row = q0 + 8 * hi + v;
      O[(size_t)(b * SL + row) * DM + hoff + nt * 16 + l15] = f2bf(o[nt][v] * l_r[v]);
    }
}

// ---------------------------------------------------------------------------
// Launcher
// ---------------------------------------------------------------------------
extern "C" void kernel_launch(void* const* d_in, const int* in_sizes, int n_in,
                              void* d_out, int out_size, void* d_ws, size_t ws_size,
                              hipStream_t stream) {
  const float* x    = (const float*)d_in[0];
  const int*   mask = (const int*)  d_in[1];
  const float* Wq   = (const float*)d_in[2];
  const float* bq   = (const float*)d_in[3];
  const float* Wk   = (const float*)d_in[4];
  const float* bk   = (const float*)d_in[5];
  const float* Wv   = (const float*)d_in[6];
  const float* bv   = (const float*)d_in[7];
  const float* Wf   = (const float*)d_in[8];
  const float* bf_  = (const float*)d_in[9];

  char* ws = (char*)d_ws;
  const size_t XB = (size_t)MT * DM * sizeof(bf16);   // 16 MB
  const size_t WB = (size_t)DM * DM * sizeof(bf16);   // 2 MB
  bf16* xh  = (bf16*)(ws);
  bf16* wqh = (bf16*)(ws + XB);
  bf16* wkh = (bf16*)(ws + XB + 1 * WB);
  bf16* wvh = (bf16*)(ws + XB + 2 * WB);
  bf16* wfh = (bf16*)(ws + XB + 3 * WB);
  bf16* Qh  = (bf16*)(ws + XB + 4 * WB);
  bf16* Kh  = (bf16*)(ws + 2 * XB + 4 * WB);
  bf16* Vh  = (bf16*)(ws + 3 * XB + 4 * WB);
  bf16* Oh  = (bf16*)(ws + 4 * XB + 4 * WB);

  const int nx = MT * DM;
  const int nw = DM * DM;
  cast_f32_bf16<<<(nx + 255) / 256, 256, 0, stream>>>(x,  xh,  nx);
  cast_f32_bf16<<<(nw + 255) / 256, 256, 0, stream>>>(Wq, wqh, nw);
  cast_f32_bf16<<<(nw + 255) / 256, 256, 0, stream>>>(Wk, wkh, nw);
  cast_f32_bf16<<<(nw + 255) / 256, 256, 0, stream>>>(Wv, wvh, nw);
  cast_f32_bf16<<<(nw + 255) / 256, 256, 0, stream>>>(Wf, wfh, nw);

  dim3 gg(DM / 128, MT / 128);                        // (8, 64)
  gemm_bf16<0><<<gg, 256, 0, stream>>>(xh, wqh, bq, Qh);
  gemm_bf16<0><<<gg, 256, 0, stream>>>(xh, wkh, bk, Kh);
  gemm_bf16<0><<<gg, 256, 0, stream>>>(xh, wvh, bv, Vh);

  attn_kernel<<<dim3(SL / 128, BS * NH), 256, 0, stream>>>(Qh, Kh, Vh, mask, Oh);

  gemm_bf16<1><<<gg, 256, 0, stream>>>(Oh, wfh, bf_, d_out);
}
